// Network_10411000726140
// MI455X (gfx1250) — matrix operand order repaired and checked by reading, not compiled, not run on hardware
//
#include <hip/hip_runtime.h>
#include <hip/hip_bf16.h>
#include <stdint.h>

#define T_LEN  1024
#define BATCH  128
#define IN_DIM 512
#define HID    1024
#define N_ACT  256
#define KTOT   (IN_DIM + HID)   // 1536
#define GATE4  (4 * HID)        // 4096

typedef __bf16 bf16_t;
typedef bf16_t v16bf __attribute__((ext_vector_type(16)));
typedef float  v8f   __attribute__((ext_vector_type(8)));
typedef unsigned int v4u  __attribute__((ext_vector_type(4)));
typedef int          v8i_t __attribute__((ext_vector_type(8)));
typedef int          v4i_t __attribute__((ext_vector_type(4)));

#if defined(__gfx1250__) && __has_builtin(__builtin_amdgcn_tensor_load_to_lds)
#define USE_TDM 1
#else
#define USE_TDM 0
#endif

union Frag16 {
    v16bf v;
    uint4 q[2];
    unsigned short u[16];
};

__device__ __forceinline__ unsigned short f2bf(float f) {
    unsigned u = __float_as_uint(f);
    unsigned r = (u + 0x7FFFu + ((u >> 16) & 1u)) >> 16;
    return (unsigned short)r;
}

__device__ __forceinline__ float sig_(float x)  { return 1.0f / (1.0f + __expf(-x)); }
__device__ __forceinline__ float tanh_(float x) { return 1.0f - 2.0f / (1.0f + __expf(2.0f * x)); }

// ---------------------------------------------------------------------------
// Prep: build bf16 combined weight Wc = [W_ih | W_hh] (4096 x 1536), bf16 W_fc,
// bsum = b_ih + b_hh, zero h0 / c0.
// ---------------------------------------------------------------------------
__global__ void prep_kernel(const float* __restrict__ W_ih, const float* __restrict__ W_hh,
                            const float* __restrict__ b_ih, const float* __restrict__ b_hh,
                            const float* __restrict__ W_fc,
                            unsigned short* __restrict__ Wc, unsigned short* __restrict__ Wfcb,
                            float* __restrict__ bsum,
                            unsigned short* __restrict__ h0, float* __restrict__ c0) {
    size_t stride = (size_t)gridDim.x * blockDim.x;
    size_t gid = (size_t)blockIdx.x * blockDim.x + threadIdx.x;
    for (size_t i = gid; i < (size_t)GATE4 * KTOT; i += stride) {
        int row = (int)(i / KTOT), k = (int)(i % KTOT);
        float v = (k < IN_DIM) ? W_ih[(size_t)row * IN_DIM + k]
                               : W_hh[(size_t)row * HID + (k - IN_DIM)];
        Wc[i] = f2bf(v);
    }
    for (size_t i = gid; i < (size_t)N_ACT * HID; i += stride) Wfcb[i] = f2bf(W_fc[i]);
    for (size_t i = gid; i < GATE4; i += stride) bsum[i] = b_ih[i] + b_hh[i];
    for (size_t i = gid; i < (size_t)BATCH * HID; i += stride) { h0[i] = 0; c0[i] = 0.0f; }
}

// ---------------------------------------------------------------------------
// One LSTM timestep: gates = [x_t, h] @ Wc^T (+bias), then pointwise update.
// Grid: 32 WGs x 256 threads. WG owns hidden cols [32*wg, 32*wg+32) and the
// four matched gate column blocks (i/f/g/o), so pointwise stays in registers.
// Weight tiles are staged into LDS by the Tensor Data Mover (one 3D-tile
// tensor_load_to_lds per 128-wide K chunk, double buffered, TENSORcnt sync).
// ---------------------------------------------------------------------------
#define JBLK   32
#define NWG    (HID / JBLK)     // 32
#define CK     128              // K-chunk staged in LDS
#define NCHUNK (KTOT / CK)      // 12
#define XCHUNK (IN_DIM / CK)    // 4
#define WROW   (CK + 8)         // padded LDS row (bf16 elems) = 272 B

__global__ __launch_bounds__(256, 1) void lstm_step_kernel(
        const float* __restrict__ x_t,              // [128][512] fp32
        const unsigned short* __restrict__ Wc,      // [4096][1536] bf16
        const float* __restrict__ bsum,             // [4096]
        const unsigned short* __restrict__ h_in,    // [128][1024] bf16
        unsigned short* __restrict__ h_out,         // [128][1024] bf16
        float* __restrict__ c_st) {                 // [128][1024] fp32
    __shared__ alignas(16) unsigned short Wl[2][4 * JBLK][WROW];  // ~68 KB

    const int tid  = threadIdx.x;
    const int wave = tid >> 5;       // 0..7  -> M-tile (batch rows 16w..16w+15)
    const int lane = tid & 31;
    const int half = lane >> 4;
    const int l15  = lane & 15;
    const int j0   = blockIdx.x * JBLK;

#if USE_TDM
    // Tensor DMA: stage one K chunk (3D tile: 128 elems x 32 rows x 4 gate
    // blocks) of this WG's gate columns into LDS buffer `buf`.
    // LDS padding: 4 DWORDs (16 B) after every 64 DWORDs (256 B = one row)
    // reproduces the WROW=136 padded row layout.
    auto tdm_stage = [&](int buf, int ch) {
        if (wave == 0) {
            unsigned long long gaddr =
                (unsigned long long)(uintptr_t)(Wc + (size_t)j0 * KTOT + ch * CK);
            unsigned lds = (unsigned)(uintptr_t)&Wl[buf][0][0];  // addr[31:0] = LDS offset

            v4u g0;
            g0[0] = 1u;                                        // count=1, user desc
            g0[1] = lds;                                       // lds_addr
            g0[2] = (unsigned)gaddr;                           // global_addr[31:0]
            g0[3] = (unsigned)((gaddr >> 32) & 0x1FFFFFFu)     // global_addr[56:32]
                    | (2u << 30);                              // type=2 ("image")

            const unsigned td0 = KTOT, td1 = GATE4;
            const unsigned tile0 = CK, tile1 = JBLK, tile2 = 4;
            const unsigned long long s0 = KTOT;                        // row stride
            const unsigned long long s1 = (unsigned long long)HID * KTOT; // gate-block stride
            v8i_t g1;
            g1[0] = (int)((1u << 16)       // data_size = 1 -> 2 bytes
                        | (1u << 20)       // pad_enable
                        | (5u << 22)       // pad_interval: 64 DWORDs (256 B)
                        | (3u << 25));     // pad_amount:   4 DWORDs (16 B)
            g1[1] = (int)((td0 & 0xFFFFu) << 16);                       // tensor_dim0 lo
            g1[2] = (int)(((td0 >> 16) & 0xFFFFu) | ((td1 & 0xFFFFu) << 16));
            g1[3] = (int)(((td1 >> 16) & 0xFFFFu) | (tile0 << 16));     // tile_dim0
            g1[4] = (int)(tile1 | (tile2 << 16));                       // tile_dim1/2
            g1[5] = (int)(s0 & 0xFFFFFFFFu);                            // dim0_stride lo
            g1[6] = (int)(((s0 >> 32) & 0xFFFFu) | ((s1 & 0xFFFFu) << 16));
            g1[7] = (int)((s1 >> 16) & 0xFFFFFFFFu);                    // dim1_stride hi

            v4i_t g2;
            g2[0] = 4; g2[1] = 0; g2[2] = 0; g2[3] = 0;                 // tensor_dim2=4
            v4i_t g3;
            g3[0] = 0; g3[1] = 0; g3[2] = 0; g3[3] = 0;
            v8i_t g4;                                                   // reserved/extra
            g4[0] = 0; g4[1] = 0; g4[2] = 0; g4[3] = 0;
            g4[4] = 0; g4[5] = 0; g4[6] = 0; g4[7] = 0;

            __builtin_amdgcn_tensor_load_to_lds(g0, g1, g2, g3, g4, 0);
        }
    };
    auto tdm_fence = [&]() {
        if (wave == 0) __builtin_amdgcn_s_wait_tensorcnt(0);
        __syncthreads();
    };
#else
    // Fallback: synchronous staging through registers.
    auto stage = [&](int buf, int ch) {
        int n_local = tid >> 1;            // 0..127
        int kh = (tid & 1) * 64;
        int g = n_local >> 5;
        int c = n_local & 31;
        int col = g * HID + j0 + c;        // gate column in [0,4096)
        const uint4* src = (const uint4*)(Wc + (size_t)col * KTOT + ch * CK + kh);
        unsigned short* dst = &Wl[buf][n_local][kh];
        #pragma unroll
        for (int i = 0; i < 8; ++i) *(uint4*)(dst + i * 8) = src[i];
    };
#endif

    // A fragment from x_t (fp32 -> bf16 in registers). ISA 16-bit A layout:
    // lane holds M = lane&15; elems 0..7 = K(8*half..), elems 8..15 = K(+16).
    auto load_a_x = [&](int kbase) -> Frag16 {
        Frag16 a;
        int m = wave * 16 + l15;
        const float* p = x_t + (size_t)m * IN_DIM + kbase + 8 * half;
        float4 f0 = *(const float4*)(p);
        float4 f1 = *(const float4*)(p + 4);
        float4 f2 = *(const float4*)(p + 16);
        float4 f3 = *(const float4*)(p + 20);
        a.u[0]  = f2bf(f0.x); a.u[1]  = f2bf(f0.y); a.u[2]  = f2bf(f0.z); a.u[3]  = f2bf(f0.w);
        a.u[4]  = f2bf(f1.x); a.u[5]  = f2bf(f1.y); a.u[6]  = f2bf(f1.z); a.u[7]  = f2bf(f1.w);
        a.u[8]  = f2bf(f2.x); a.u[9]  = f2bf(f2.y); a.u[10] = f2bf(f2.z); a.u[11] = f2bf(f2.w);
        a.u[12] = f2bf(f3.x); a.u[13] = f2bf(f3.y); a.u[14] = f2bf(f3.z); a.u[15] = f2bf(f3.w);
        return a;
    };

    // A fragment from h (already bf16).
    auto load_a_h = [&](int kbase) -> Frag16 {
        Frag16 a;
        int m = wave * 16 + l15;
        const unsigned short* p = h_in + (size_t)m * HID + kbase + 8 * half;
        a.q[0] = *(const uint4*)(p);
        a.q[1] = *(const uint4*)(p + 16);
        return a;
    };

    // B fragment from LDS. ISA 16-bit B layout: lane holds N = lane&15,
    // elems 0..7 = K(8*half..), elems 8..15 = K(+16).
    auto load_b = [&](int buf, int nt, int kk) -> Frag16 {
        Frag16 b;
        const unsigned short* p = &Wl[buf][nt * 16 + l15][kk * 32 + 8 * half];
        b.q[0] = *(const uint4*)(p);
        b.q[1] = *(const uint4*)(p + 16);
        return b;
    };

    v8f zero = {0.f, 0.f, 0.f, 0.f, 0.f, 0.f, 0.f, 0.f};
    v8f acc[8];
    #pragma unroll
    for (int i = 0; i < 8; ++i) acc[i] = zero;

#if USE_TDM
    tdm_stage(0, 0);
    tdm_fence();
#else
    stage(0, 0);
    __syncthreads();
#endif
    for (int ch = 0; ch < NCHUNK; ++ch) {
        int cur = ch & 1;
#if USE_TDM
        if (ch + 1 < NCHUNK) tdm_stage(cur ^ 1, ch + 1);
#else
        if (ch + 1 < NCHUNK) stage(cur ^ 1, ch + 1);
#endif
        if (ch < XCHUNK) {
            #pragma unroll
            for (int kk = 0; kk < 4; ++kk) {
                Frag16 a = load_a_x(ch * CK + kk * 32);
                #pragma unroll
                for (int nt = 0; nt < 8; ++nt) {
                    Frag16 b = load_b(cur, nt, kk);
                    acc[nt] = __builtin_amdgcn_wmma_f32_16x16x32_bf16(
                        false, a.v, false, b.v, (short)0, acc[nt], false, false);
                }
            }
        } else {
            #pragma unroll
            for (int kk = 0; kk < 4; ++kk) {
                Frag16 a = load_a_h(ch * CK - IN_DIM + kk * 32);
                #pragma unroll
                for (int nt = 0; nt < 8; ++nt) {
                    Frag16 b = load_b(cur, nt, kk);
                    acc[nt] = __builtin_amdgcn_wmma_f32_16x16x32_bf16(
                        false, a.v, false, b.v, (short)0, acc[nt], false, false);
                }
            }
        }
#if USE_TDM
        tdm_fence();
#else
        __syncthreads();
#endif
    }

    // Pointwise LSTM update straight from the accumulators.
    // acc[0..1]=i, acc[2..3]=f, acc[4..5]=g, acc[6..7]=o (p = column half).
    #pragma unroll
    for (int p = 0; p < 2; ++p) {
        int j = j0 + p * 16 + l15;
        float bi = bsum[0 * HID + j];
        float bf = bsum[1 * HID + j];
        float bg = bsum[2 * HID + j];
        float bo = bsum[3 * HID + j];
        #pragma unroll
        for (int r = 0; r < 8; ++r) {
            int m = wave * 16 + half * 8 + r;           // C/D layout: M = 8*half + r
            float iv = acc[0 + p][r] + bi;
            float fv = acc[2 + p][r] + bf;
            float gv = acc[4 + p][r] + bg;
            float ov = acc[6 + p][r] + bo;
            size_t idx = (size_t)m * HID + j;
            float cold = c_st[idx];
            float cnew = sig_(fv) * cold + sig_(iv) * tanh_(gv);
            float hnew = sig_(ov) * tanh_(cnew);
            c_st[idx]  = cnew;
            h_out[idx] = f2bf(hnew);
        }
    }
}

// ---------------------------------------------------------------------------
// FC (1024 -> 256) + row-wise log_softmax for one timestep.
// Grid: 8 WGs x 256 threads; WG handles 16 batch rows, waves split 256 cols.
// ---------------------------------------------------------------------------
__global__ __launch_bounds__(256, 1) void fc_logsoftmax_kernel(
        const unsigned short* __restrict__ h_in,  // bf16 [128][1024]
        const unsigned short* __restrict__ Wfc,   // bf16 [256][1024]
        const float* __restrict__ b_fc,           // [256]
        float* __restrict__ out_t) {              // [128][256]
    __shared__ alignas(16) unsigned short Al[16][HID + 8];  // ~33 KB
    __shared__ float logits[16][N_ACT];                     // 16 KB
    __shared__ float pred[16][16];
    __shared__ float rowred[16];

    const int tid  = threadIdx.x;
    const int wave = tid >> 5;
    const int lane = tid & 31;
    const int half = lane >> 4;
    const int l15  = lane & 15;
    const int m0   = blockIdx.x * 16;

    // Stage the 16 h rows for this WG into LDS (shared A across all 8 waves).
    {
        int rr = tid >> 4;
        int ks = (tid & 15) * 64;
        const uint4* src = (const uint4*)(h_in + (size_t)(m0 + rr) * HID + ks);
        uint4* dst = (uint4*)(&Al[rr][ks]);
        #pragma unroll
        for (int i = 0; i < 8; ++i) dst[i] = src[i];
    }
    __syncthreads();

    v8f zero = {0.f, 0.f, 0.f, 0.f, 0.f, 0.f, 0.f, 0.f};
    v8f acc[2];
    acc[0] = zero; acc[1] = zero;
    const int ncol0 = wave * 32;

    for (int k = 0; k < HID; k += 32) {
        Frag16 a;
        const unsigned short* ap = &Al[l15][k + 8 * half];
        a.q[0] = *(const uint4*)(ap);
        a.q[1] = *(const uint4*)(ap + 16);
        #pragma unroll
        for (int nt = 0; nt < 2; ++nt) {
            Frag16 b;
            const unsigned short* bp = Wfc + (size_t)(ncol0 + nt * 16 + l15) * HID + k + 8 * half;
            b.q[0] = *(const uint4*)(bp);
            b.q[1] = *(const uint4*)(bp + 16);
            acc[nt] = __builtin_amdgcn_wmma_f32_16x16x32_bf16(
                false, a.v, false, b.v, (short)0, acc[nt], false, false);
        }
    }

    // Dump logits (+bias) to LDS for the row reductions.
    #pragma unroll
    for (int nt = 0; nt < 2; ++nt) {
        int n = ncol0 + nt * 16 + l15;
        float bias = b_fc[n];
        #pragma unroll
        for (int r = 0; r < 8; ++r) logits[half * 8 + r][n] = acc[nt][r] + bias;
    }
    __syncthreads();

    int row = tid >> 4;
    int seg = tid & 15;
    float mx = -3.0e38f;
    #pragma unroll
    for (int i = 0; i < 16; ++i) mx = fmaxf(mx, logits[row][seg * 16 + i]);
    pred[row][seg] = mx;
    __syncthreads();
    if (seg == 0) {
        float m2 = pred[row][0];
        #pragma unroll
        for (int i = 1; i < 16; ++i) m2 = fmaxf(m2, pred[row][i]);
        rowred[row] = m2;
    }
    __syncthreads();
    float rmax = rowred[row];
    float s = 0.f;
    #pragma unroll
    for (int i = 0; i < 16; ++i) s += __expf(logits[row][seg * 16 + i] - rmax);
    __syncthreads();   // everyone has read rmax before rowred is overwritten
    pred[row][seg] = s;
    __syncthreads();
    if (seg == 0) {
        float s2 = 0.f;
        #pragma unroll
        for (int i = 0; i < 16; ++i) s2 += pred[row][i];
        rowred[row] = rmax + __logf(s2);
    }
    __syncthreads();
    float lz = rowred[row];
    float* op = out_t + (size_t)(m0 + row) * N_ACT + seg * 16;
    #pragma unroll
    for (int i = 0; i < 16; i += 4) {
        float4 v;
        v.x = logits[row][seg * 16 + i + 0] - lz;
        v.y = logits[row][seg * 16 + i + 1] - lz;
        v.z = logits[row][seg * 16 + i + 2] - lz;
        v.w = logits[row][seg * 16 + i + 3] - lz;
        *(float4*)(op + i) = v;
    }
}

// ---------------------------------------------------------------------------
// Host-side launcher. Workspace layout (≈13.6 MB):
//   Wc   bf16 [4096][1536]   12,582,912 B
//   Wfc  bf16 [ 256][1024]      524,288 B
//   bsum f32  [4096]             16,384 B
//   hbuf bf16 [2][128][1024]    524,288 B   (double-buffered h)
//   c    f32  [128][1024]       524,288 B
// ---------------------------------------------------------------------------
extern "C" void kernel_launch(void* const* d_in, const int* in_sizes, int n_in,
                              void* d_out, int out_size, void* d_ws, size_t ws_size,
                              hipStream_t stream) {
    const float* x    = (const float*)d_in[0];
    const float* W_ih = (const float*)d_in[1];
    const float* W_hh = (const float*)d_in[2];
    const float* b_ih = (const float*)d_in[3];
    const float* b_hh = (const float*)d_in[4];
    const float* W_fc = (const float*)d_in[5];
    const float* b_fc = (const float*)d_in[6];
    float* out = (float*)d_out;

    char* ws = (char*)d_ws;
    size_t off = 0;
    unsigned short* Wc   = (unsigned short*)(ws + off); off += (size_t)GATE4 * KTOT * 2;
    unsigned short* Wfcb = (unsigned short*)(ws + off); off += (size_t)N_ACT * HID * 2;
    float*          bsum = (float*)(ws + off);          off += (size_t)GATE4 * 4;
    unsigned short* hbuf = (unsigned short*)(ws + off); off += (size_t)2 * BATCH * HID * 2;
    float*          cst  = (float*)(ws + off);          off += (size_t)BATCH * HID * 4;

    prep_kernel<<<2048, 256, 0, stream>>>(W_ih, W_hh, b_ih, b_hh, W_fc,
                                          Wc, Wfcb, bsum, hbuf, cst);

    for (int t = 0; t < T_LEN; ++t) {
        const unsigned short* hin = hbuf + (size_t)(t & 1) * BATCH * HID;
        unsigned short* hout      = hbuf + (size_t)((t + 1) & 1) * BATCH * HID;
        lstm_step_kernel<<<NWG, 256, 0, stream>>>(
            x + (size_t)t * BATCH * IN_DIM, Wc, bsum, hin, hout, cst);
        fc_logsoftmax_kernel<<<8, 256, 0, stream>>>(
            hout, Wfcb, b_fc, out + (size_t)t * BATCH * N_ACT);
    }
}
